// FalconAttention_55327768708228
// MI455X (gfx1250) — matrix operand order repaired and checked by reading, not compiled, not run on hardware
//
#include <hip/hip_runtime.h>

// ---------------------------------------------------------------------------
// Falcon MQA block for MI455X (gfx1250, wave32, WMMA).
// Strategy: fp32 inputs -> bf16 once; all GEMM work on
// v_wmma_f32_16x16x32_bf16 with fp32 accumulation; softmax in fp32.
// ~105 GFLOP total, working set < L2 (192MB) => compute bound => bf16 WMMA.
// Attention K-tile staging uses CDNA5 async global->LDS copies (ASYNCcnt).
// ---------------------------------------------------------------------------

#define DEV __device__ __forceinline__

typedef __attribute__((ext_vector_type(16))) __bf16 v16bf;
typedef __attribute__((ext_vector_type(8)))  float  v8f;

struct alignas(16) U128 { unsigned int w[4]; };
union Frag  { U128 u[2]; v16bf v; };
union Pack8 { U128 u; unsigned short s[8]; };

static constexpr int B_  = 2;
static constexpr int S_  = 2048;
static constexpr int HID = 2048;
static constexpr int NH  = 32;
static constexpr int HD  = 64;
static constexpr int NQKV = HID + 2 * HD;   // 2176

DEV unsigned short f2bf(float f) {
  union { float f; unsigned u; } x; x.f = f;
  unsigned r = x.u + 0x7FFFu + ((x.u >> 16) & 1u);   // round-to-nearest-even
  return (unsigned short)(r >> 16);
}

DEV v8f wmma_bf16(const Frag& a, const Frag& b, v8f c) {
  return __builtin_amdgcn_wmma_f32_16x16x32_bf16(
      /*neg_a=*/false, a.v, /*neg_b=*/false, b.v,
      /*c_mod=*/(short)0, c, /*reuse_a=*/false, /*reuse_b=*/false);
}

// Async global->LDS b128 copy (CDNA5, tracked by ASYNCcnt).
// ldsoff = LDS byte offset (low 32 bits of flat shared address), g = global src.
DEV void async_copy_b128(unsigned ldsoff, const void* g) {
  asm volatile("global_load_async_to_lds_b128 %0, %1, off"
               :: "v"(ldsoff), "v"(g) : "memory");
}
DEV void wait_async0() {
  asm volatile("s_wait_asynccnt 0x0" ::: "memory");
}

// ---------------------------------------------------------------------------
// Kernel 1: fp32 -> bf16 bulk convert
// ---------------------------------------------------------------------------
__global__ void k_f32_to_bf16(const float* __restrict__ in,
                              unsigned short* __restrict__ out, size_t n) {
  size_t i = (size_t)blockIdx.x * blockDim.x + threadIdx.x;
  if (i < n) out[i] = f2bf(in[i]);
}

// ---------------------------------------------------------------------------
// Kernel 2/5: C(MxN, f32) = A(MxK, bf16) * W(NxK, bf16)^T
// 128x128 block tile, 256 threads = 8 waves, wave tile 32x64, K-step 32.
// LDS rows padded to 40 elems (80B, 16B aligned) => fragment loads are b128.
// ---------------------------------------------------------------------------
__global__ __launch_bounds__(256) void k_gemm_bf16(
    const unsigned short* __restrict__ A,
    const unsigned short* __restrict__ W,
    float* __restrict__ C, int M, int N, int K)
{
  __shared__ unsigned short As[128 * 40];
  __shared__ unsigned short Ws[128 * 40];
  const int tid  = threadIdx.x;
  const int wv   = tid >> 5, lane = tid & 31;
  const int hf   = lane >> 4, l16 = lane & 15;
  const int bm   = blockIdx.x * 128, bn = blockIdx.y * 128;
  const int wm   = (wv >> 1) * 32;   // 4 waves down M
  const int wn   = (wv & 1) * 64;    // 2 waves across N

  v8f zf = {0.f,0.f,0.f,0.f,0.f,0.f,0.f,0.f};
  v8f acc[2][4];
  #pragma unroll
  for (int i = 0; i < 2; ++i)
    #pragma unroll
    for (int j = 0; j < 4; ++j) acc[i][j] = zf;

  for (int k0 = 0; k0 < K; k0 += 32) {
    // cooperative stage: 512 16B-chunks per tile, 2 per thread per tile
    #pragma unroll
    for (int i = tid; i < 512; i += 256) {
      const int row = i >> 2, ck = (i & 3) * 8;
      *(U128*)&As[row * 40 + ck] =
          *(const U128*)&A[(size_t)(bm + row) * K + k0 + ck];
      *(U128*)&Ws[row * 40 + ck] =
          *(const U128*)&W[(size_t)(bn + row) * K + k0 + ck];
    }
    __syncthreads();

    Frag af[2], bf[4];
    #pragma unroll
    for (int i = 0; i < 2; ++i) {   // A 16x32: e<8 -> K=hf*8+e ; e>=8 -> 16+hf*8+(e-8)
      const unsigned short* p = &As[(wm + i * 16 + l16) * 40 + hf * 8];
      af[i].u[0] = *(const U128*)p;
      af[i].u[1] = *(const U128*)(p + 16);
    }
    #pragma unroll
    for (int j = 0; j < 4; ++j) {   // B 32x16: e<8 -> K=hf*8+e ; e>=8 -> 16+hf*8+(e-8), col = l16
      const unsigned short* p = &Ws[(wn + j * 16 + l16) * 40 + hf * 8];
      bf[j].u[0] = *(const U128*)p;
      bf[j].u[1] = *(const U128*)(p + 16);
    }
    #pragma unroll
    for (int i = 0; i < 2; ++i)
      #pragma unroll
      for (int j = 0; j < 4; ++j)
        acc[i][j] = wmma_bf16(af[i], bf[j], acc[i][j]);
    __syncthreads();
  }

  #pragma unroll
  for (int i = 0; i < 2; ++i)
    #pragma unroll
    for (int j = 0; j < 4; ++j)
      #pragma unroll
      for (int r = 0; r < 8; ++r) {
        const int row = bm + wm + i * 16 + r + hf * 8;   // C layout: M=r+hf*8
        const int col = bn + wn + j * 16 + l16;          //           N=l16
        C[(size_t)row * N + col] = acc[i][j][r];
      }
}

// ---------------------------------------------------------------------------
// Kernel 3: RoPE + split fused QKV (fp32) into q/k/v (bf16).
// q: [b][h][s][d] with 1/sqrt(HD) folded in; k,v: [b][s][d].
// ---------------------------------------------------------------------------
__global__ void k_rope_split(const float* __restrict__ fused,
                             unsigned short* __restrict__ q,
                             unsigned short* __restrict__ k,
                             unsigned short* __restrict__ v)
{
  const size_t idx = (size_t)blockIdx.x * blockDim.x + threadIdx.x;
  const size_t total = (size_t)B_ * S_ * (NH + 2) * HD;
  if (idx >= total) return;
  const int d = (int)(idx & 63);
  size_t t = idx >> 6;
  const int h = (int)(t % (NH + 2));
  const size_t bs = t / (NH + 2);          // b*S + s
  const int s = (int)(bs % S_);
  const int b = (int)(bs / S_);

  const float* row = fused + bs * NQKV + (size_t)h * HD;
  float x = row[d];
  if (h < NH + 1) {                        // rope on q heads and k
    const int i = d & 31;
    const float freq = __expf(-(float)i * 0.2878231366242558f); // ln(1e4)/32
    const float ang = (float)s * freq;
    const float c = cosf(ang), sn = sinf(ang);
    const float other = (d < 32) ? -row[d + 32] : row[d - 32];
    x = x * c + other * sn;
  }
  if (h < NH) {
    x *= 0.125f;                           // fold 1/sqrt(64) into q
    q[((size_t)(b * NH + h) * S_ + s) * HD + d] = f2bf(x);
  } else if (h == NH) {
    k[bs * HD + d] = f2bf(x);
  } else {
    v[bs * HD + d] = f2bf(x);
  }
}

// ---------------------------------------------------------------------------
// Kernel 4: causal flash attention, multi-query (one K/V head per batch).
// Block = 64 query rows (4 waves x 16), 64-key tiles staged in LDS.
// K tile staged with global_load_async_to_lds_b128 (ASYNCcnt), V tile staged
// transposed (b128 global reads, scalar LDS scatter).
// ---------------------------------------------------------------------------
__global__ __launch_bounds__(128) void k_attn(
    const unsigned short* __restrict__ Q,   // [B][NH][S][HD]
    const unsigned short* __restrict__ Kg,  // [B][S][HD]
    const unsigned short* __restrict__ Vg,  // [B][S][HD]
    unsigned short* __restrict__ ctx)       // [B][S][NH*HD] bf16
{
  __shared__ unsigned short Ks[64 * 72];        // [key][d]
  __shared__ unsigned short Vs[64 * 72];        // transposed: [d][key]
  __shared__ unsigned short Ps[4][16 * 72];     // per-wave P scratch
  const int tid = threadIdx.x;
  const int wv  = tid >> 5, lane = tid & 31;
  const int hf  = lane >> 4, l16 = lane & 15;
  const int q0  = blockIdx.x * 64;
  const int h   = blockIdx.y;
  const int b   = blockIdx.z;
  const int qw  = q0 + wv * 16;                 // this wave's 16 query rows

  // Q tile 16x64 as two 16x32 A-fragments, held in registers for whole loop
  const unsigned short* qbase = Q + ((size_t)(b * NH + h) * S_ + qw) * HD;
  Frag aq[2];
  #pragma unroll
  for (int kk = 0; kk < 2; ++kk) {
    const unsigned short* p = qbase + l16 * HD + kk * 32 + hf * 8;
    aq[kk].u[0] = *(const U128*)p;
    aq[kk].u[1] = *(const U128*)(p + 16);
  }

  v8f zf = {0.f,0.f,0.f,0.f,0.f,0.f,0.f,0.f};
  v8f acc[4]; float m[8], l[8];
  #pragma unroll
  for (int n = 0; n < 4; ++n) acc[n] = zf;
  #pragma unroll
  for (int r = 0; r < 8; ++r) { m[r] = -1e30f; l[r] = 0.f; }

  const unsigned short* kb = Kg + (size_t)b * S_ * HD;
  const unsigned short* vb = Vg + (size_t)b * S_ * HD;
  const int jmax = q0 / 64;                     // causal block limit

  for (int j = 0; j <= jmax; ++j) {
    const int key0 = j * 64;
    // stage K with CDNA5 async global->LDS copies (no VGPR round-trip)
    #pragma unroll
    for (int i = tid; i < 512; i += 128) {
      const int row = i >> 3, ck = (i & 7) * 8;
      const unsigned ldsoff = (unsigned)(size_t)&Ks[row * 72 + ck];
      async_copy_b128(ldsoff, &kb[(size_t)(key0 + row) * HD + ck]);
    }
    // stage V transposed: b128 global reads, 8 scalar LDS scatter stores each
    #pragma unroll
    for (int c = tid; c < 512; c += 128) {
      const int key = c >> 3, d0 = (c & 7) * 8;
      Pack8 pk;
      pk.u = *(const U128*)&vb[(size_t)(key0 + key) * HD + d0];
      #pragma unroll
      for (int t = 0; t < 8; ++t)
        Vs[(d0 + t) * 72 + key] = pk.s[t];
    }
    wait_async0();          // our async K chunks are in LDS
    __syncthreads();        // everyone's staging visible

    // scores S = Q * K^T  (4 tiles of 16x16, each = 2 WMMAs over d)
    v8f sc[4];
    #pragma unroll
    for (int n = 0; n < 4; ++n) {
      Frag bk[2];
      #pragma unroll
      for (int kk = 0; kk < 2; ++kk) {
        const unsigned short* p = &Ks[(n * 16 + l16) * 72 + kk * 32 + hf * 8];
        bk[kk].u[0] = *(const U128*)p;
        bk[kk].u[1] = *(const U128*)(p + 16);
      }
      v8f c = zf;
      c = wmma_bf16(aq[0], bk[0], c);
      c = wmma_bf16(aq[1], bk[1], c);
      sc[n] = c;
    }

    // causal mask + online softmax (row M = r + hf*8; 16-lane half reduction)
    float rmax[8];
    #pragma unroll
    for (int r = 0; r < 8; ++r) rmax[r] = -1e30f;
    #pragma unroll
    for (int n = 0; n < 4; ++n) {
      const int kg = key0 + n * 16 + l16;
      #pragma unroll
      for (int r = 0; r < 8; ++r) {
        const int qg = qw + r + hf * 8;
        float s = sc[n][r];
        if (kg > qg) s = -1e30f;
        sc[n][r] = s;
        rmax[r] = fmaxf(rmax[r], s);
      }
    }
    #pragma unroll
    for (int r = 0; r < 8; ++r) {
      rmax[r] = fmaxf(rmax[r], __shfl_xor(rmax[r], 1, 32));
      rmax[r] = fmaxf(rmax[r], __shfl_xor(rmax[r], 2, 32));
      rmax[r] = fmaxf(rmax[r], __shfl_xor(rmax[r], 4, 32));
      rmax[r] = fmaxf(rmax[r], __shfl_xor(rmax[r], 8, 32));
    }
    float alpha[8], rsum[8];
    #pragma unroll
    for (int r = 0; r < 8; ++r) {
      const float mn = fmaxf(m[r], rmax[r]);
      alpha[r] = __expf(m[r] - mn);
      m[r] = mn; rsum[r] = 0.f;
    }
    #pragma unroll
    for (int n = 0; n < 4; ++n)
      #pragma unroll
      for (int r = 0; r < 8; ++r) {
        const float p = __expf(sc[n][r] - m[r]);
        sc[n][r] = p;
        rsum[r] += p;
      }
    #pragma unroll
    for (int r = 0; r < 8; ++r) {
      rsum[r] += __shfl_xor(rsum[r], 1, 32);
      rsum[r] += __shfl_xor(rsum[r], 2, 32);
      rsum[r] += __shfl_xor(rsum[r], 4, 32);
      rsum[r] += __shfl_xor(rsum[r], 8, 32);
      l[r] = l[r] * alpha[r] + rsum[r];
    }
    #pragma unroll
    for (int n = 0; n < 4; ++n)
      #pragma unroll
      for (int r = 0; r < 8; ++r)
        acc[n][r] = acc[n][r] * alpha[r];

    // P (C-layout) -> per-wave LDS scratch -> A-layout fragments.
    // No barrier needed: same-wave LDS ops are in-order, waits auto-inserted.
    unsigned short* pw = Ps[wv];
    #pragma unroll
    for (int n = 0; n < 4; ++n)
      #pragma unroll
      for (int r = 0; r < 8; ++r)
        pw[(r + hf * 8) * 72 + n * 16 + l16] = f2bf(sc[n][r]);

    Frag ap[2];
    #pragma unroll
    for (int kk = 0; kk < 2; ++kk) {
      const unsigned short* p = &pw[l16 * 72 + kk * 32 + hf * 8];
      ap[kk].u[0] = *(const U128*)p;
      ap[kk].u[1] = *(const U128*)(p + 16);
    }
    #pragma unroll
    for (int n = 0; n < 4; ++n) {
      Frag bv[2];
      #pragma unroll
      for (int kk = 0; kk < 2; ++kk) {
        const unsigned short* p = &Vs[(n * 16 + l16) * 72 + kk * 32 + hf * 8];
        bv[kk].u[0] = *(const U128*)p;
        bv[kk].u[1] = *(const U128*)(p + 16);
      }
      acc[n] = wmma_bf16(ap[0], bv[0], acc[n]);
      acc[n] = wmma_bf16(ap[1], bv[1], acc[n]);
    }
    __syncthreads();   // Ks/Vs about to be overwritten
  }

  // epilogue: ctx[b][s][h*64 + d] = acc / l  (bf16 for the dense GEMM)
  #pragma unroll
  for (int n = 0; n < 4; ++n)
    #pragma unroll
    for (int r = 0; r < 8; ++r) {
      const float o = acc[n][r] / l[r];
      const int srow = qw + r + hf * 8;
      ctx[((size_t)b * S_ + srow) * (NH * HD) + h * HD + n * 16 + l16] = f2bf(o);
    }
}

// ---------------------------------------------------------------------------
// Host-side orchestration
// ---------------------------------------------------------------------------
extern "C" void kernel_launch(void* const* d_in, const int* in_sizes, int n_in,
                              void* d_out, int out_size, void* d_ws, size_t ws_size,
                              hipStream_t stream) {
  const float* hidden  = (const float*)d_in[0];   // [B][S][HID] fp32
  const float* w_qkv   = (const float*)d_in[1];   // [2176][2048] fp32
  const float* w_dense = (const float*)d_in[2];   // [2048][2048] fp32
  float* out = (float*)d_out;                     // [B][S][HID] fp32

  const size_t n_hid  = (size_t)B_ * S_ * HID;        // 8388608
  const size_t n_wqkv = (size_t)NQKV * HID;           // 4456448
  const size_t n_wd   = (size_t)HID * HID;            // 4194304

  char* ws = (char*)d_ws;
  unsigned short* hid_bf  = (unsigned short*)(ws);                 // 16 MiB
  unsigned short* wqkv_bf = (unsigned short*)(ws + 16777216);      // 8.5 MiB
  unsigned short* wd_bf   = (unsigned short*)(ws + 25690112);      // 8 MiB
  float*          fused   = (float*)        (ws + 34078720);       // 34 MiB
  unsigned short* q_bf    = (unsigned short*)(ws + 69730304);      // 16 MiB
  unsigned short* k_bf    = (unsigned short*)(ws + 86507520);      // 0.5 MiB
  unsigned short* v_bf    = (unsigned short*)(ws + 87031808);      // 0.5 MiB
  unsigned short* ctx_bf  = (unsigned short*)(ws + 87556096);      // 16 MiB

  // 1) fp32 -> bf16
  k_f32_to_bf16<<<(unsigned)((n_hid  + 255) / 256), 256, 0, stream>>>(hidden,  hid_bf,  n_hid);
  k_f32_to_bf16<<<(unsigned)((n_wqkv + 255) / 256), 256, 0, stream>>>(w_qkv,   wqkv_bf, n_wqkv);
  k_f32_to_bf16<<<(unsigned)((n_wd   + 255) / 256), 256, 0, stream>>>(w_dense, wd_bf,   n_wd);

  // 2) fused = hidden @ w_qkv^T : (4096 x 2176 x 2048)
  {
    dim3 grid((B_ * S_) / 128, NQKV / 128);   // 32 x 17
    k_gemm_bf16<<<grid, 256, 0, stream>>>(hid_bf, wqkv_bf, fused,
                                          B_ * S_, NQKV, HID);
  }

  // 3) RoPE + split
  {
    const size_t total = (size_t)B_ * S_ * (NH + 2) * HD;
    k_rope_split<<<(unsigned)((total + 255) / 256), 256, 0, stream>>>(
        fused, q_bf, k_bf, v_bf);
  }

  // 4) causal flash attention -> ctx (bf16)
  {
    dim3 grid(S_ / 64, NH, B_);               // 32 x 32 x 2
    k_attn<<<grid, 128, 0, stream>>>(q_bf, k_bf, v_bf, ctx_bf);
  }

  // 5) out = ctx @ w_dense^T : (4096 x 2048 x 2048)
  {
    dim3 grid((B_ * S_) / 128, HID / 128);    // 32 x 16
    k_gemm_bf16<<<grid, 256, 0, stream>>>(ctx_bf, wd_bf, out,
                                          B_ * S_, HID, HID);
  }
}
